// AttentionModule_61924838474357
// MI455X (gfx1250) — compile-verified, matrix operand order checked by reading, not run on hardware
//
#include <hip/hip_runtime.h>
#include <hip/hip_bf16.h>
#include <stdint.h>

// ---------------------------------------------------------------------------
// Problem sizes (fixed by the reference)
// ---------------------------------------------------------------------------
#define N_TOK 8192
#define D_IN  1024
#define D_HD  128
#define KB    64          // keys per flash-attention block iteration
#define SPLITS 4          // key-range splits (flash-decoding style)
#define QK_SLAB 64        // K-slab width for the QKV GEMM

// LDS pitches (elements): rows padded by 8 elems (TDM pad: 4 dwords) so the
// 16 lanes of a half-wave reading 16B chunks stride across all 64 LDS banks.
#define K_PITCH 136       // 128 + 8 elems = 272 B rows
#define V_PITCH 72        //  64 + 8 elems = 144 B rows
#define P_PITCH 72        //  64 + 8 elems = 144 B rows
#define S_PITCH 72        // qkv-gemm slab rows (64 + 8 elems)

typedef __attribute__((ext_vector_type(16))) __bf16 v16bf;
typedef __attribute__((ext_vector_type(8)))  float  v8f;
typedef __attribute__((ext_vector_type(4)))  uint32_t tdm_v4u;
typedef __attribute__((ext_vector_type(4)))  int      tdm_v4i;
typedef __attribute__((ext_vector_type(8)))  int      tdm_v8i;

// fp32 -> bf16, round-to-nearest-even
__device__ __forceinline__ uint16_t f2bf(float f) {
    uint32_t u = __builtin_bit_cast(uint32_t, f);
    u += 0x7FFFu + ((u >> 16) & 1u);
    return (uint16_t)(u >> 16);
}

union FragU { v16bf v; uint4 q[2]; };

// 16x32 bf16 WMMA operand fragment (A and B patterns are identical):
// lane l16 owns row/col l16; half (hl) selects K-chunks kc+8*hl and kc+16+8*hl.
__device__ __forceinline__ v16bf load_frag(const uint16_t* rowPtr, int kc, int hl) {
    FragU u;
    u.q[0] = *reinterpret_cast<const uint4*>(rowPtr + kc + 8 * hl);
    u.q[1] = *reinterpret_cast<const uint4*>(rowPtr + kc + 16 + 8 * hl);
    return u.v;
}

__device__ __forceinline__ v8f wmma_bf16(v16bf a, v16bf b, v8f c) {
    return __builtin_amdgcn_wmma_f32_16x16x32_bf16(
        false, a, false, b, (short)0, c, false, false);
}

// ---------------------------------------------------------------------------
// Tensor Data Mover: 2D tile load (global -> LDS), bf16 elements.
// D# packing per CDNA5 ISA. workgroup_mask = 0
// (not in a cluster). pad_* inserts 4 dwords per row -> conflict-free pitch.
// ---------------------------------------------------------------------------
__device__ __forceinline__ void tdm_load_2d(uint32_t lds_off, const void* gaddr,
                                            uint32_t tdim0, uint32_t tdim1,
                                            uint32_t tile0, uint32_t tile1,
                                            uint32_t stride0,
                                            uint32_t pad_interval_code) {
    uint64_t ga = (uint64_t)(uintptr_t)gaddr;
    tdm_v4u g0;
    g0[0] = 1u;                                   // count=1, user mode
    g0[1] = lds_off;                              // lds_addr (bytes)
    g0[2] = (uint32_t)ga;                         // global_addr[31:0]
    g0[3] = (uint32_t)((ga >> 32) & 0x01FFFFFFu)  // global_addr[56:32]
          | (2u << 30);                           // type = 2 ("image")
    tdm_v8i g1;
    g1[0] = (int)((1u << 16)                      // data_size = 2 bytes
          | (1u << 20)                            // pad_enable
          | (pad_interval_code << 22)             // dwords-per-row code
          | (3u << 25));                          // pad_amount = 4 dwords
    g1[1] = (int)((tdim0 & 0xFFFFu) << 16);
    g1[2] = (int)(((tdim0 >> 16) & 0xFFFFu) | ((tdim1 & 0xFFFFu) << 16));
    g1[3] = (int)(((tdim1 >> 16) & 0xFFFFu) | ((tile0 & 0xFFFFu) << 16));
    g1[4] = (int)(tile1 & 0xFFFFu);               // tile_dim1; tile_dim2=0
    g1[5] = (int)stride0;                         // tensor_dim0_stride[31:0]
    g1[6] = 0;                                    // stride0 hi / dim1_stride lo
    g1[7] = 0;
    tdm_v4i z4 = {0, 0, 0, 0};
#if defined(__clang_major__) && (__clang_major__ >= 23)
    tdm_v8i z8 = {0, 0, 0, 0, 0, 0, 0, 0};
    __builtin_amdgcn_tensor_load_to_lds(g0, g1, z4, z4, z8, 0);
#else
    __builtin_amdgcn_tensor_load_to_lds(g0, g1, z4, z4, 0);
#endif
}

__device__ __forceinline__ uint32_t lds_offset(const void* p) {
    return (uint32_t)(uintptr_t)p;                // generic addr low 32 = LDS offset
}

// ---------------------------------------------------------------------------
// Pass 0: fp32 -> bf16 conversion (x4 vectorized)
// ---------------------------------------------------------------------------
__global__ void cvt_f32_to_bf16(const float* __restrict__ src,
                                uint16_t* __restrict__ dst, int n4) {
    int i = blockIdx.x * blockDim.x + threadIdx.x;
    if (i >= n4) return;
    float4 f = reinterpret_cast<const float4*>(src)[i];
    uint2 o;
    o.x = (uint32_t)f2bf(f.x) | ((uint32_t)f2bf(f.y) << 16);
    o.y = (uint32_t)f2bf(f.z) | ((uint32_t)f2bf(f.w) << 16);
    reinterpret_cast<uint2*>(dst)[i] = o;
}

// ---------------------------------------------------------------------------
// Pass 1: q/k/v = x @ W^T, TDM-staged. Each block owns 128 rows x one matrix.
// K processed in 64-wide slabs: wave 0 DMAs x-slab (128x64) and W-slab
// (128x64) into double-buffered padded LDS while 8 waves compute from the
// other buffer (16 WMMAs/slab/wave). Removes the load->wait0->wmma serial
// pattern: global latency is absorbed by the async TDM engine.
// ---------------------------------------------------------------------------
__global__ void qkv_gemm(const uint16_t* __restrict__ xb,
                         const uint16_t* __restrict__ wq,
                         const uint16_t* __restrict__ wk,
                         const uint16_t* __restrict__ wv,
                         uint16_t* __restrict__ qb,
                         uint16_t* __restrict__ kb,
                         uint16_t* __restrict__ vT) {
    __shared__ __align__(16) uint16_t xs[2][128 * S_PITCH + 8];
    __shared__ __align__(16) uint16_t wsh[2][128 * S_PITCH + 8];

    const int lane = threadIdx.x & 31;
    const int wave = threadIdx.x >> 5;
    const int hl   = lane >> 4;
    const int l16  = lane & 15;
    const int mBlk = blockIdx.x * 128;            // 128 rows per block
    const int m0   = mBlk + wave * 16;            // this wave's 16 rows
    const int mat  = blockIdx.y;
    const uint16_t* W = (mat == 0) ? wq : (mat == 1) ? wk : wv;
    const bool issuer = (wave == 0);

    // Prologue: DMA slab 0 of x rows and W into buffer 0
    if (issuer) {
        tdm_load_2d(lds_offset(&xs[0][0]), xb + (size_t)mBlk * D_IN,
                    D_IN, N_TOK, QK_SLAB, 128, D_IN, 4u /* 32 dw rows */);
        tdm_load_2d(lds_offset(&wsh[0][0]), W,
                    D_IN, D_HD, QK_SLAB, 128, D_IN, 4u);
        __builtin_amdgcn_s_wait_tensorcnt(0);
    }
    __syncthreads();

    v8f acc[8] = {};
    const int nSlab = D_IN / QK_SLAB;             // 16
    for (int sl = 0; sl < nSlab; ++sl) {
        const int buf = sl & 1;
        if (issuer && (sl + 1) < nSlab) {
            const int kn = (sl + 1) * QK_SLAB;
            tdm_load_2d(lds_offset(&xs[buf ^ 1][0]), xb + (size_t)mBlk * D_IN + kn,
                        D_IN, N_TOK, QK_SLAB, 128, D_IN, 4u);
            tdm_load_2d(lds_offset(&wsh[buf ^ 1][0]), W + kn,
                        D_IN, D_HD, QK_SLAB, 128, D_IN, 4u);
        }

        const uint16_t* aRow = &xs[buf][(wave * 16 + l16) * S_PITCH];
        #pragma unroll
        for (int c = 0; c < 2; ++c) {             // two 32-wide chunks per slab
            v16bf a = load_frag(aRow, c * 32, hl);
            #pragma unroll
            for (int nt = 0; nt < 8; ++nt) {
                v16bf b = load_frag(&wsh[buf][(nt * 16 + l16) * S_PITCH], c * 32, hl);
                acc[nt] = wmma_bf16(a, b, acc[nt]);
            }
        }

        __syncthreads();                          // done reading buf
        if (issuer && (sl + 1) < nSlab)
            __builtin_amdgcn_s_wait_tensorcnt(0);
        __syncthreads();                          // next buffer ready
    }

    if (mat < 2) {
        uint16_t* outp = (mat == 0) ? qb : kb;
        #pragma unroll
        for (int nt = 0; nt < 8; ++nt)
            #pragma unroll
            for (int r = 0; r < 8; ++r)
                outp[(size_t)(m0 + r + 8 * hl) * D_HD + nt * 16 + l16] =
                    f2bf(acc[nt][r]);
    } else {
        // vT[n][m]: each lane's 8 C values are 8 consecutive rows of column n
        #pragma unroll
        for (int nt = 0; nt < 8; ++nt) {
            int n = nt * 16 + l16;
            uint4 o;
            o.x = (uint32_t)f2bf(acc[nt][0]) | ((uint32_t)f2bf(acc[nt][1]) << 16);
            o.y = (uint32_t)f2bf(acc[nt][2]) | ((uint32_t)f2bf(acc[nt][3]) << 16);
            o.z = (uint32_t)f2bf(acc[nt][4]) | ((uint32_t)f2bf(acc[nt][5]) << 16);
            o.w = (uint32_t)f2bf(acc[nt][6]) | ((uint32_t)f2bf(acc[nt][7]) << 16);
            *reinterpret_cast<uint4*>(vT + (size_t)n * N_TOK + m0 + 8 * hl) = o;
        }
    }
}

// ---------------------------------------------------------------------------
// Pass 2: flash attention with TDM-staged, double-buffered LDS k/v tiles and
// split-K partials. One wave owns 16 query rows; 64-key steps; 32 WMMAs/step.
// Partials stored unnormalized with per-row running (m, l) in log2 domain.
// ---------------------------------------------------------------------------
__global__ void flash_attn(const uint16_t* __restrict__ qb,
                           const uint16_t* __restrict__ kmat,
                           const uint16_t* __restrict__ vT,
                           float* __restrict__ accP,
                           float* __restrict__ mP,
                           float* __restrict__ lP) {
    __shared__ __align__(16) uint16_t kTile[2][KB * K_PITCH + 8];
    __shared__ __align__(16) uint16_t vTile[2][D_HD * V_PITCH + 8];
    __shared__ __align__(16) uint16_t pshm[8][16 * P_PITCH];

    const int lane = threadIdx.x & 31;
    const int wave = threadIdx.x >> 5;
    const int hl   = lane >> 4;
    const int l16  = lane & 15;
    const int m0   = (blockIdx.x * 8 + wave) * 16;
    const int split = blockIdx.y;
    const int keyBegin = split * (N_TOK / SPLITS);
    const int nIter    = (N_TOK / SPLITS) / KB;
    const bool issuer  = (wave == 0);
    uint16_t* myP = &pshm[wave][0];

    // Resident q fragments (head dim 128 = 4 K-chunks)
    const uint16_t* qRow = qb + (size_t)(m0 + l16) * D_HD;
    v16bf qf[4];
    #pragma unroll
    for (int c = 0; c < 4; ++c) qf[c] = load_frag(qRow, c * 32, hl);

    // Prologue: DMA the first k/v tiles into buffer 0
    if (issuer) {
        tdm_load_2d(lds_offset(&kTile[0][0]), kmat + (size_t)keyBegin * D_HD,
                    D_HD, N_TOK, D_HD, KB, D_HD, 5u /* 64 dw rows */);
        tdm_load_2d(lds_offset(&vTile[0][0]), vT + keyBegin,
                    N_TOK, D_HD, KB, D_HD, N_TOK, 4u /* 32 dw rows */);
        __builtin_amdgcn_s_wait_tensorcnt(0);
    }
    __syncthreads();

    v8f acc[8] = {};
    float mrow[8], lrow[8];
    #pragma unroll
    for (int r = 0; r < 8; ++r) { mrow[r] = -3.0e38f; lrow[r] = 0.0f; }
    const float t = 11.3137084989847603f * 1.4426950408889634f; // sqrt(128)*log2e

    for (int it = 0; it < nIter; ++it) {
        const int buf = it & 1;
        if (issuer && (it + 1) < nIter) {
            const int keyN = keyBegin + (it + 1) * KB;
            tdm_load_2d(lds_offset(&kTile[buf ^ 1][0]), kmat + (size_t)keyN * D_HD,
                        D_HD, N_TOK, D_HD, KB, D_HD, 5u);
            tdm_load_2d(lds_offset(&vTile[buf ^ 1][0]), vT + keyN,
                        N_TOK, D_HD, KB, D_HD, N_TOK, 4u);
        }

        // ---- S = q @ k^T for 64 keys: 4 fragments x 4 K-chunks ----
        v8f s[4] = {};
        #pragma unroll
        for (int f = 0; f < 4; ++f) {
            const uint16_t* kr = &kTile[buf][(f * 16 + l16) * K_PITCH];
            #pragma unroll
            for (int c = 0; c < 4; ++c)
                s[f] = wmma_bf16(qf[c], load_frag(kr, c * 32, hl), s[f]);
        }

        // ---- online softmax in log2 domain ----
        float e[4][8], mloc[8];
        #pragma unroll
        for (int r = 0; r < 8; ++r) {
            mloc[r] = -3.0e38f;
            #pragma unroll
            for (int f = 0; f < 4; ++f) {
                e[f][r] = s[f][r] * t;
                mloc[r] = fmaxf(mloc[r], e[f][r]);
            }
        }
        #pragma unroll
        for (int d = 8; d >= 1; d >>= 1)
            #pragma unroll
            for (int r = 0; r < 8; ++r)
                mloc[r] = fmaxf(mloc[r], __shfl_xor(mloc[r], d, 32));

        float psum[8];
        #pragma unroll
        for (int r = 0; r < 8; ++r) {
            float mnew  = fmaxf(mrow[r], mloc[r]);
            float alpha = __builtin_amdgcn_exp2f(mrow[r] - mnew);
            lrow[r] *= alpha;
            #pragma unroll
            for (int nt = 0; nt < 8; ++nt) acc[nt][r] *= alpha;
            psum[r] = 0.0f;
            #pragma unroll
            for (int f = 0; f < 4; ++f) {
                float p = __builtin_amdgcn_exp2f(e[f][r] - mnew);
                psum[r] += p;
                myP[(r + 8 * hl) * P_PITCH + f * 16 + l16] = f2bf(p);
            }
            mrow[r] = mnew;
        }
        #pragma unroll
        for (int d = 8; d >= 1; d >>= 1)
            #pragma unroll
            for (int r = 0; r < 8; ++r)
                psum[r] += __shfl_xor(psum[r], d, 32);
        #pragma unroll
        for (int r = 0; r < 8; ++r) lrow[r] += psum[r];

        // ---- reload P as two A fragments (K = 64) ----
        FragU u0, u1;
        u0.q[0] = *reinterpret_cast<const uint4*>(myP + l16 * P_PITCH + 8 * hl);
        u0.q[1] = *reinterpret_cast<const uint4*>(myP + l16 * P_PITCH + 16 + 8 * hl);
        u1.q[0] = *reinterpret_cast<const uint4*>(myP + l16 * P_PITCH + 32 + 8 * hl);
        u1.q[1] = *reinterpret_cast<const uint4*>(myP + l16 * P_PITCH + 48 + 8 * hl);

        // ---- acc += P @ V (vTile rows are K-contiguous local keys) ----
        #pragma unroll
        for (int nt = 0; nt < 8; ++nt) {
            const uint16_t* vr = &vTile[buf][(nt * 16 + l16) * V_PITCH];
            acc[nt] = wmma_bf16(u0.v, load_frag(vr, 0, hl), acc[nt]);
            acc[nt] = wmma_bf16(u1.v, load_frag(vr, 32, hl), acc[nt]);
        }

        __syncthreads();                       // everyone done reading buf
        if (issuer && (it + 1) < nIter)
            __builtin_amdgcn_s_wait_tensorcnt(0);
        __syncthreads();                       // next buffer visible to all
    }

    // ---- epilogue: store unnormalized partial + per-row (m, l) ----
    #pragma unroll
    for (int nt = 0; nt < 8; ++nt)
        #pragma unroll
        for (int r = 0; r < 8; ++r)
            accP[((size_t)split * N_TOK + m0 + r + 8 * hl) * D_HD + nt * 16 + l16] =
                acc[nt][r];
    if (l16 == 0) {
        #pragma unroll
        for (int r = 0; r < 8; ++r) {
            int m = m0 + r + 8 * hl;
            mP[split * N_TOK + m] = mrow[r];
            lP[split * N_TOK + m] = lrow[r];
        }
    }
}

// ---------------------------------------------------------------------------
// Pass 3: merge split-K partials: out = sum_s w_s*acc_s / sum_s w_s*l_s,
// w_s = exp2(m_s - max_s m_s). One thread per (row, 4 cols).
// ---------------------------------------------------------------------------
__global__ void merge_splits(const float* __restrict__ accP,
                             const float* __restrict__ mP,
                             const float* __restrict__ lP,
                             float* __restrict__ out) {
    int idx = blockIdx.x * blockDim.x + threadIdx.x;   // 8192 * 32
    int m  = idx >> 5;
    int n4 = (idx & 31) * 4;
    float M = -3.0e38f;
    #pragma unroll
    for (int s = 0; s < SPLITS; ++s) M = fmaxf(M, mP[s * N_TOK + m]);
    float L = 0.0f;
    float4 o = make_float4(0.f, 0.f, 0.f, 0.f);
    #pragma unroll
    for (int s = 0; s < SPLITS; ++s) {
        float w = __builtin_amdgcn_exp2f(mP[s * N_TOK + m] - M);
        L += w * lP[s * N_TOK + m];
        float4 a = *reinterpret_cast<const float4*>(
            &accP[((size_t)s * N_TOK + m) * D_HD + n4]);
        o.x += w * a.x; o.y += w * a.y; o.z += w * a.z; o.w += w * a.w;
    }
    float inv = 1.0f / L;
    o.x *= inv; o.y *= inv; o.z *= inv; o.w *= inv;
    *reinterpret_cast<float4*>(&out[(size_t)m * D_HD + n4]) = o;
}

// ---------------------------------------------------------------------------
// Host-side launcher
// ---------------------------------------------------------------------------
extern "C" void kernel_launch(void* const* d_in, const int* in_sizes, int n_in,
                              void* d_out, int out_size, void* d_ws, size_t ws_size,
                              hipStream_t stream) {
    (void)in_sizes; (void)n_in; (void)out_size; (void)ws_size;
    const float* x  = (const float*)d_in[0];
    const float* Wq = (const float*)d_in[1];
    const float* Wk = (const float*)d_in[2];
    const float* Wv = (const float*)d_in[3];
    float* out = (float*)d_out;

    uint8_t* ws = (uint8_t*)d_ws;
    size_t off = 0;
    auto take = [&](size_t bytes) -> void* {
        void* p = ws + off;
        off += (bytes + 255) & ~(size_t)255;
        return p;
    };
    uint16_t* xb  = (uint16_t*)take((size_t)N_TOK * D_IN * 2);   // 16 MB
    uint16_t* wqb = (uint16_t*)take((size_t)D_HD * D_IN * 2);
    uint16_t* wkb = (uint16_t*)take((size_t)D_HD * D_IN * 2);
    uint16_t* wvb = (uint16_t*)take((size_t)D_HD * D_IN * 2);
    uint16_t* qb  = (uint16_t*)take((size_t)N_TOK * D_HD * 2);   // 2 MB
    uint16_t* kb  = (uint16_t*)take((size_t)N_TOK * D_HD * 2);   // 2 MB
    uint16_t* vT  = (uint16_t*)take((size_t)N_TOK * D_HD * 2);   // 2 MB
    float* accP   = (float*)take((size_t)SPLITS * N_TOK * D_HD * 4); // 16.8 MB
    float* mP     = (float*)take((size_t)SPLITS * N_TOK * 4);
    float* lP     = (float*)take((size_t)SPLITS * N_TOK * 4);

    int n4x = N_TOK * D_IN / 4;
    cvt_f32_to_bf16<<<(n4x + 255) / 256, 256, 0, stream>>>(x, xb, n4x);
    int n4w = D_HD * D_IN / 4;
    cvt_f32_to_bf16<<<(n4w + 255) / 256, 256, 0, stream>>>(Wq, wqb, n4w);
    cvt_f32_to_bf16<<<(n4w + 255) / 256, 256, 0, stream>>>(Wk, wkb, n4w);
    cvt_f32_to_bf16<<<(n4w + 255) / 256, 256, 0, stream>>>(Wv, wvb, n4w);

    qkv_gemm<<<dim3(64, 3), 256, 0, stream>>>(xb, wqb, wkb, wvb, qb, kb, vT);

    flash_attn<<<dim3(64, SPLITS), 256, 0, stream>>>(qb, kb, vT, accP, mP, lP);

    merge_splits<<<(N_TOK * 32) / 256, 256, 0, stream>>>(accP, mP, lP, out);
}
